// MultiScaleTemporalPooling_36842229465406
// MI455X (gfx1250) — matrix operand order repaired and checked by reading, not compiled, not run on hardware
//
#include <hip/hip_runtime.h>
#include <hip/hip_bf16.h>

typedef __bf16 bf16_t;
typedef __attribute__((ext_vector_type(16))) __bf16 v16bf;
typedef __attribute__((ext_vector_type(8)))  __bf16 v8bf;
typedef __attribute__((ext_vector_type(8)))  float  v8f;

#define DIM   512
#define TT    64
#define NB    1024
#define NROWS (NB*TT)        // 65536
#define NTILES (NROWS/16)    // 4096
#define KT    16             // 512 / 32 k-steps
#define HSTR  516            // padded LDS row stride (4 mod 64 banks -> conflict-free column reads)

// ---------------------------------------------------------------------------
// exact GELU (erf form, matches approximate=False)
__device__ __forceinline__ float gelu_exact(float v) {
    return 0.5f * v * (1.0f + erff(v * 0.70710678118654752440f));
}

// ---------------------------------------------------------------------------
// Pack fp32 weights into bf16 WMMA B-fragment layout.
// B (32x16 per k-step): lane l holds N = l%16; elements e=0..15 hold
// K = kstep*32 + (l<16 ? e : 16+e).  Packed so GEMM loads 32 contiguous B/lane.
__global__ void __launch_bounds__(256)
pack_weights(const float* __restrict__ W1_0, const float* __restrict__ W1_1,
             const float* __restrict__ W1_2, const float* __restrict__ Wp,
             bf16_t* __restrict__ pW1, bf16_t* __restrict__ pWp) {
    int gid = blockIdx.x * blockDim.x + threadIdx.x;
    const int W1_GROUPS = 3 * 16 * 16 * 32;     // scorers * ntiles * ktiles * lanes
    const int WP_GROUPS = 3 * 32 * 16 * 32;     // parts   * ntiles * ktiles * lanes
    if (gid < W1_GROUPS) {
        int l  = gid & 31;
        int kt = (gid >> 5) & 15;
        int nt = (gid >> 9) & 15;
        int i  = gid >> 13;
        const float* W = (i == 0) ? W1_0 : ((i == 1) ? W1_1 : W1_2);  // [512,256] row-major (k,n)
        int n     = nt * 16 + (l & 15);
        int kbase = kt * 32 + ((l < 16) ? 0 : 16);
        bf16_t* dst = pW1 + (size_t)gid * 16;
        #pragma unroll
        for (int e = 0; e < 16; e++) dst[e] = (bf16_t)W[(size_t)(kbase + e) * 256 + n];
    } else {
        int g = gid - W1_GROUPS;
        if (g >= WP_GROUPS) return;
        int l  = g & 31;
        int kt = (g >> 5) & 15;
        int nt = (g >> 9) & 31;
        int p  = g >> 14;                        // Wp part: rows [p*512, p*512+512)
        int n     = nt * 16 + (l & 15);
        int kbase = kt * 32 + ((l < 16) ? 0 : 16);
        bf16_t* dst = pWp + (size_t)g * 16;
        #pragma unroll
        for (int e = 0; e < 16; e++)
            dst[e] = (bf16_t)Wp[(size_t)(p * 512 + kbase + e) * 512 + n];
    }
}

// ---------------------------------------------------------------------------
// Stage a 16x512 fp32 tile (optionally row-scaled) as bf16 A-fragments in LDS.
// A (16x32 per k-step): lane l: M = l%16; half=l/16; elements 0..15 = K kbase..+15
// where kbase = kt*32 + half*16 (matches packed B order).
__device__ __forceinline__ void stage_A_f32(const float* __restrict__ rowbase,
                                            float rscale, bf16_t* Alds) {
    int l = threadIdx.x & 31;
    int m = l & 15, half = l >> 4;
    const float* rp = rowbase + (size_t)m * DIM;
    #pragma unroll
    for (int kt = 0; kt < KT; kt++) {
        int k0 = kt * 32 + half * 16;
        float4 f0 = *(const float4*)(rp + k0);
        float4 f1 = *(const float4*)(rp + k0 + 4);
        float4 f2 = *(const float4*)(rp + k0 + 8);
        float4 f3 = *(const float4*)(rp + k0 + 12);
        v16bf v;
        v[0]  = (bf16_t)(f0.x * rscale); v[1]  = (bf16_t)(f0.y * rscale);
        v[2]  = (bf16_t)(f0.z * rscale); v[3]  = (bf16_t)(f0.w * rscale);
        v[4]  = (bf16_t)(f1.x * rscale); v[5]  = (bf16_t)(f1.y * rscale);
        v[6]  = (bf16_t)(f1.z * rscale); v[7]  = (bf16_t)(f1.w * rscale);
        v[8]  = (bf16_t)(f2.x * rscale); v[9]  = (bf16_t)(f2.y * rscale);
        v[10] = (bf16_t)(f2.z * rscale); v[11] = (bf16_t)(f2.w * rscale);
        v[12] = (bf16_t)(f3.x * rscale); v[13] = (bf16_t)(f3.y * rscale);
        v[14] = (bf16_t)(f3.z * rscale); v[15] = (bf16_t)(f3.w * rscale);
        *(v16bf*)(Alds + kt * DIM + l * 16) = v;
    }
}

__device__ __forceinline__ void stage_A_bf16(const bf16_t* __restrict__ rowbase,
                                             bf16_t* Alds) {
    int l = threadIdx.x & 31;
    int m = l & 15, half = l >> 4;
    const bf16_t* rp = rowbase + (size_t)m * DIM;
    #pragma unroll
    for (int kt = 0; kt < KT; kt++) {
        int k0 = kt * 32 + half * 16;
        v8bf lo = *(const v8bf*)(rp + k0);
        v8bf hi = *(const v8bf*)(rp + k0 + 8);
        v8bf* d = (v8bf*)(Alds + kt * DIM + l * 16);
        d[0] = lo; d[1] = hi;
    }
}

// 16x16 output tile = A(16x512, LDS) x B(512x16, packed global), fp32 accum.
__device__ __forceinline__ v8f gemm_tile(const bf16_t* Alds,
                                         const bf16_t* __restrict__ Bbase) {
    int l = threadIdx.x & 31;
    v8f c = {};
    #pragma unroll
    for (int kt = 0; kt < KT; kt++) {
        v16bf a = *(const v16bf*)(Alds + kt * DIM + l * 16);
        v16bf b = *(const v16bf*)(Bbase + (size_t)kt * DIM + l * 16);
        c = __builtin_amdgcn_wmma_f32_16x16x32_bf16(false, a, false, b,
                                                    (short)0, c, false, false);
    }
    return c;
}

// ---------------------------------------------------------------------------
// Scorers: s_i = GELU(x@W1_i + b1_i) @ W2_i + b2_i for i=0,1,2.
// One wave per 16-row tile; A staged once, reused by all 3 scorers (48 tiles).
__global__ void __launch_bounds__(32)
scorer_kernel(const float* __restrict__ x, const bf16_t* __restrict__ pW1,
              const float* b1_0, const float* W2_0, const float* b2_0,
              const float* b1_1, const float* W2_1, const float* b2_1,
              const float* b1_2, const float* W2_2, const float* b2_2,
              float* __restrict__ sbuf) {
    __shared__ bf16_t Alds[16 * DIM];
    int row0 = blockIdx.x * 16;
    int l = threadIdx.x;
    stage_A_f32(x + (size_t)row0 * DIM, 1.0f, Alds);
    __syncthreads();
    const float* b1s[3] = {b1_0, b1_1, b1_2};
    const float* W2s[3] = {W2_0, W2_1, W2_2};
    const float* b2s[3] = {b2_0, b2_1, b2_2};
    #pragma unroll
    for (int i = 0; i < 3; i++) {
        float racc[8] = {0, 0, 0, 0, 0, 0, 0, 0};
        for (int nt = 0; nt < 16; nt++) {
            const bf16_t* Bb = pW1 + (size_t)(i * 16 + nt) * 16 * DIM;
            v8f c = gemm_tile(Alds, Bb);
            int n = nt * 16 + (l & 15);
            float b1v = b1s[i][n];
            float w2v = W2s[i][n];
            #pragma unroll
            for (int r = 0; r < 8; r++)
                racc[r] += gelu_exact(c[r] + b1v) * w2v;
        }
        // reduce the 16 N-lanes within each half (halves hold rows r and r+8)
        #pragma unroll
        for (int r = 0; r < 8; r++) {
            float v = racc[r];
            v += __shfl_xor(v, 8, 32);
            v += __shfl_xor(v, 4, 32);
            v += __shfl_xor(v, 2, 32);
            v += __shfl_xor(v, 1, 32);
            racc[r] = v;
        }
        if ((l & 15) == 0) {
            float b2v = b2s[i][0];
            int rbase = row0 + (l >> 4) * 8;
            #pragma unroll
            for (int r = 0; r < 8; r++)
                sbuf[(size_t)i * NROWS + rbase + r] = racc[r] + b2v;
        }
    }
}

// ---------------------------------------------------------------------------
// Softmax weights + attention pooling: pooled0[b,d], pooled1[b,g,d] (bf16).
__global__ void __launch_bounds__(256)
pool_kernel(const float* __restrict__ x, const float* __restrict__ sbuf,
            bf16_t* __restrict__ pooled0, bf16_t* __restrict__ pooled1) {
    __shared__ float w0[TT], w1[TT];
    int b = blockIdx.x;
    int tid = threadIdx.x;
    if (tid < TT) {
        const float* s0 = sbuf + (size_t)b * TT;              // scorer 0
        float mx = -3.4e38f;
        for (int t = 0; t < TT; t++) mx = fmaxf(mx, s0[t]);
        float den = 0.0f;
        for (int t = 0; t < TT; t++) den += expf(s0[t] - mx);
        w0[tid] = expf(s0[tid] - mx) / den;
        const float* s1 = sbuf + (size_t)NROWS + (size_t)b * TT;  // scorer 1
        int g0 = tid & ~7;
        float mx1 = -3.4e38f;
        for (int t = 0; t < 8; t++) mx1 = fmaxf(mx1, s1[g0 + t]);
        float d1 = 0.0f;
        for (int t = 0; t < 8; t++) d1 += expf(s1[g0 + t] - mx1);
        w1[tid] = expf(s1[tid] - mx1) / d1;
    }
    __syncthreads();
    const float* xb = x + (size_t)b * TT * DIM;
    for (int d = tid; d < DIM; d += 256) {
        float a0 = 0.0f;
        #pragma unroll
        for (int g = 0; g < 8; g++) {
            float ag = 0.0f;
            #pragma unroll
            for (int s = 0; s < 8; s++) {
                int t = g * 8 + s;
                float xv = xb[(size_t)t * DIM + d];
                a0 += w0[t] * xv;
                ag += w1[t] * xv;
            }
            pooled1[((size_t)b * 8 + g) * DIM + d] = (bf16_t)ag;
        }
        pooled0[(size_t)b * DIM + d] = (bf16_t)a0;
    }
}

// ---------------------------------------------------------------------------
// c0 = pooled0 @ Wp[0:512,:]  (64 tiles), c1 = pooled1 @ Wp[512:1024,:] (512 tiles)
__global__ void __launch_bounds__(32)
poolproj_kernel(const bf16_t* __restrict__ pooled0, const bf16_t* __restrict__ pooled1,
                const bf16_t* __restrict__ pWp, float* __restrict__ c0,
                float* __restrict__ c1) {
    __shared__ bf16_t Alds[16 * DIM];
    int blk = blockIdx.x;
    const bf16_t* Asrc;
    const bf16_t* Bsrc;
    float* Cdst;
    if (blk < 64) {
        Asrc = pooled0 + (size_t)blk * 16 * DIM;
        Cdst = c0 + (size_t)blk * 16 * DIM;
        Bsrc = pWp;                                       // part 0
    } else {
        int t = blk - 64;
        Asrc = pooled1 + (size_t)t * 16 * DIM;
        Cdst = c1 + (size_t)t * 16 * DIM;
        Bsrc = pWp + (size_t)1 * 32 * 16 * DIM;           // part 1
    }
    stage_A_bf16(Asrc, Alds);
    __syncthreads();
    int l = threadIdx.x;
    for (int nt = 0; nt < 32; nt++) {
        v8f c = gemm_tile(Alds, Bsrc + (size_t)nt * 16 * DIM);
        int col = nt * 16 + (l & 15);
        #pragma unroll
        for (int r = 0; r < 8; r++)
            Cdst[(size_t)(r + 8 * (l >> 4)) * DIM + col] = c[r];
    }
}

// ---------------------------------------------------------------------------
// Final: h = GELU((x*sigmoid(s2))@Wp2 + c0[b] + c1[b,g] + bp); out = LayerNorm(h)
__global__ void __launch_bounds__(32)
final_kernel(const float* __restrict__ x, const float* __restrict__ sbuf,
             const bf16_t* __restrict__ pWp, const float* __restrict__ c0,
             const float* __restrict__ c1, const float* __restrict__ bp,
             const float* __restrict__ ln_g, const float* __restrict__ ln_b,
             float* __restrict__ out) {
    __shared__ bf16_t Alds[16 * DIM];
    __shared__ float hbuf[16 * HSTR];
    __shared__ float mu_s[16], rs_s[16];
    int row0 = blockIdx.x * 16;
    int l = threadIdx.x;
    int b  = row0 >> 6;
    int t0 = row0 & 63;
    int m = l & 15, half = l >> 4;
    float s2v = sbuf[(size_t)2 * NROWS + row0 + m];
    float sig = 1.0f / (1.0f + expf(-s2v));
    stage_A_f32(x + (size_t)row0 * DIM, sig, Alds);
    __syncthreads();
    const bf16_t* B2 = pWp + (size_t)2 * 32 * 16 * DIM;   // part 2
    for (int nt = 0; nt < 32; nt++) {
        v8f c = gemm_tile(Alds, B2 + (size_t)nt * 16 * DIM);
        int col = nt * 16 + (l & 15);
        float c0v = c0[(size_t)b * DIM + col];
        float bpv = bp[col];
        #pragma unroll
        for (int r = 0; r < 8; r++) {
            int rr = r + 8 * half;
            int g = (t0 + rr) >> 3;
            float v = c[r] + c0v + c1[((size_t)b * 8 + g) * DIM + col] + bpv;
            hbuf[rr * HSTR + col] = gelu_exact(v);
        }
    }
    __syncthreads();
    {   // per-row mean/var: 2 lanes per row, 256 cols each, combine via xor-16
        int r = l & 15;
        const float* hp = hbuf + r * HSTR + half * 256;
        float s = 0.0f, sq = 0.0f;
        for (int c = 0; c < 256; c++) { float v = hp[c]; s += v; sq += v * v; }
        s  += __shfl_xor(s, 16, 32);
        sq += __shfl_xor(sq, 16, 32);
        float mu  = s * (1.0f / 512.0f);
        float var = sq * (1.0f / 512.0f) - mu * mu;
        if (l < 16) { mu_s[r] = mu; rs_s[r] = rsqrtf(var + 1e-5f); }
    }
    __syncthreads();
    float* ob = out + (size_t)row0 * DIM;
    for (int r = 0; r < 16; r++) {
        float mu = mu_s[r], rs = rs_s[r];
        for (int c = l; c < DIM; c += 32)
            ob[(size_t)r * DIM + c] = (hbuf[r * HSTR + c] - mu) * rs * ln_g[c] + ln_b[c];
    }
}

// ---------------------------------------------------------------------------
extern "C" void kernel_launch(void* const* d_in, const int* in_sizes, int n_in,
                              void* d_out, int out_size, void* d_ws, size_t ws_size,
                              hipStream_t stream) {
    const float* x    = (const float*)d_in[0];
    const float* W1_0 = (const float*)d_in[1];
    const float* b1_0 = (const float*)d_in[2];
    const float* W2_0 = (const float*)d_in[3];
    const float* b2_0 = (const float*)d_in[4];
    const float* W1_1 = (const float*)d_in[5];
    const float* b1_1 = (const float*)d_in[6];
    const float* W2_1 = (const float*)d_in[7];
    const float* b2_1 = (const float*)d_in[8];
    const float* W1_2 = (const float*)d_in[9];
    const float* b1_2 = (const float*)d_in[10];
    const float* W2_2 = (const float*)d_in[11];
    const float* b2_2 = (const float*)d_in[12];
    const float* Wp   = (const float*)d_in[13];
    const float* bp   = (const float*)d_in[14];
    const float* ln_g = (const float*)d_in[15];
    const float* ln_b = (const float*)d_in[16];
    float* out = (float*)d_out;

    char* ws = (char*)d_ws;
    size_t off = 0;
    bf16_t* pW1     = (bf16_t*)(ws + off); off += (size_t)3 * 512 * 256 * 2;   // 768 KB
    bf16_t* pWp     = (bf16_t*)(ws + off); off += (size_t)1536 * 512 * 2;      // 1.5 MB
    float*  sbuf    = (float*)(ws + off);  off += (size_t)3 * NROWS * 4;       // 768 KB
    bf16_t* pooled0 = (bf16_t*)(ws + off); off += (size_t)NB * 512 * 2;        // 1 MB
    bf16_t* pooled1 = (bf16_t*)(ws + off); off += (size_t)NB * 8 * 512 * 2;    // 8 MB
    float*  c0      = (float*)(ws + off);  off += (size_t)NB * 512 * 4;        // 2 MB
    float*  c1      = (float*)(ws + off);  off += (size_t)NB * 8 * 512 * 4;    // 16 MB

    pack_weights<<<(3 * 16 * 16 * 32 + 3 * 32 * 16 * 32 + 255) / 256, 256, 0, stream>>>(
        W1_0, W1_1, W1_2, Wp, pW1, pWp);
    scorer_kernel<<<NTILES, 32, 0, stream>>>(
        x, pW1, b1_0, W2_0, b2_0, b1_1, W2_1, b2_1, b1_2, W2_2, b2_2, sbuf);
    pool_kernel<<<NB, 256, 0, stream>>>(x, sbuf, pooled0, pooled1);
    poolproj_kernel<<<64 + 512, 32, 0, stream>>>(pooled0, pooled1, pWp, c0, c1);
    final_kernel<<<NTILES, 32, 0, stream>>>(x, sbuf, pWp, c0, c1, bp, ln_g, ln_b, out);
}
